// SATUpsampler_v2_22488448762526
// MI455X (gfx1250) — compile-verified, matrix operand order checked by reading, not run on hardware
//
#include <hip/hip_runtime.h>
#include <hip/hip_bf16.h>

// ---------------------------------------------------------------------------
// SATUpsampler for MI455X (gfx1250, wave32, WMMA 16x16x32 f16 -> f32)
// B=4, C=64, H=W=128, S=4, K=5, HR = 512x512
// ---------------------------------------------------------------------------

typedef _Float16 v16h __attribute__((ext_vector_type(16)));
typedef _Float16 h8   __attribute__((ext_vector_type(8)));
typedef float    v8f  __attribute__((ext_vector_type(8)));

#define WMMA_F16(a, b, c) \
    __builtin_amdgcn_wmma_f32_16x16x32_f16(false, (a), false, (b), (short)0, (c), false, false)

__device__ __forceinline__ float geluf(float v) {
    return 0.5f * v * (1.0f + erff(v * 0.70710678118654752f));
}
__device__ __forceinline__ int refl128(int p) {  // jnp reflect pad, H=W=128
    p = (p < 0) ? -p : p;
    return (p > 127) ? (254 - p) : p;
}

// ---------------- workspace layout (bytes, 256-aligned blocks) -------------
static constexpr size_t OFF_H1   = 0;                       // 4*64*128*128 f16
static constexpr size_t OFF_H2   = OFF_H1   + 8388608;      // f16
static constexpr size_t OFF_XH   = OFF_H2   + 8388608;      // x as f16
static constexpr size_t OFF_FILT = OFF_XH   + 8388608;      // 4*16*25*16384 f16
static constexpr size_t OFF_M1X  = OFF_FILT + 52428800;     // 4*64*16384 f32
static constexpr size_t OFF_GATE = OFF_M1X  + 16777216;     // 4*16384 f32
static constexpr size_t OFF_WPC  = OFF_GATE + 262144;       // 4*18*32*16 f16
static constexpr size_t OFF_WPP  = OFF_WPC  + 73728;        // 16*2*2*32*16 f16
static constexpr size_t OFF_WPM1 = OFF_WPP  + 65536;        // 4*2*32*16 f16
static constexpr size_t OFF_WPM2 = OFF_WPM1 + 8192;         // 4*2*32*16 f16

// ------------------------ weight pack kernels ------------------------------
// A-fragment per-lane layout (ISA 7.12.2, 16-bit A 16x32):
//   lane L: row M = L%16 ; j<8 -> Klocal = j + 8*(L>=16) ; j>=8 -> Klocal = j+8 + 8*(L>=16)

__global__ void k_pack_conv(const float* __restrict__ w, _Float16* __restrict__ o) {
    int idx = blockIdx.x * 256 + threadIdx.x;           // 4*18*32*16 = 36864
    if (idx >= 4 * 18 * 32 * 16) return;
    int j = idx & 15, lane = (idx >> 4) & 31;
    int kc = (idx >> 9) % 18, t = (idx >> 9) / 18;
    int hi = lane >> 4;
    int M = t * 16 + (lane & 15);
    int kl = ((j < 8) ? j : j + 8) + 8 * hi;
    int Kg = kc * 32 + kl;                              // K = tap*64 + cin
    int tap = Kg >> 6, cin = Kg & 63;
    o[idx] = (_Float16)w[((M * 64 + cin) * 3 + tap / 3) * 3 + tap % 3];
}

__global__ void k_pack_proj(const float* __restrict__ w, _Float16* __restrict__ o) {
    int idx = blockIdx.x * 256 + threadIdx.x;           // 16*2*2*32*16 = 32768
    if (idx >= 32768) return;
    int j = idx & 15, lane = (idx >> 4) & 31;
    int kc = (idx >> 9) & 1, tile = (idx >> 10) & 1, s = idx >> 11;
    int hi = lane >> 4;
    int rip = tile * 16 + (lane & 15);                  // row in phase, valid < 25
    int kl = ((j < 8) ? j : j + 8) + 8 * hi;
    int cin = kc * 32 + kl;
    float v = (rip < 25) ? w[(s * 25 + rip) * 64 + cin] : 0.0f;
    o[idx] = (_Float16)v;
}

__global__ void k_pack64(const float* __restrict__ w, int ldw, _Float16* __restrict__ o) {
    int idx = blockIdx.x * 256 + threadIdx.x;           // 4*2*32*16 = 4096
    if (idx >= 4096) return;
    int j = idx & 15, lane = (idx >> 4) & 31;
    int kc = (idx >> 9) & 1, t = idx >> 10;
    int hi = lane >> 4;
    int M = t * 16 + (lane & 15);
    int kl = ((j < 8) ? j : j + 8) + 8 * hi;
    o[idx] = (_Float16)w[M * ldw + kc * 32 + kl];
}

__global__ void k_x2h(const float* __restrict__ x, _Float16* __restrict__ o) {
    int idx = blockIdx.x * 256 + threadIdx.x;           // 4,194,304 exact
    o[idx] = (_Float16)x[idx];
}

// ------------------ stage 1: depthwise 3x3 (SAME, zero) + GELU -------------
__global__ void k_dw_gelu(const float* __restrict__ x, const float* __restrict__ wdw,
                          _Float16* __restrict__ h1) {
    int idx = blockIdx.x * 256 + threadIdx.x;           // 4,194,304 exact
    int xx = idx & 127, yy = (idx >> 7) & 127, c = (idx >> 14) & 63, b = idx >> 20;
    const float* w = wdw + c * 9;
    float acc = 0.0f;
#pragma unroll
    for (int ky = 0; ky < 3; ++ky) {
        int gy = yy + ky - 1;
        if ((unsigned)gy >= 128u) continue;
#pragma unroll
        for (int kx = 0; kx < 3; ++kx) {
            int gx = xx + kx - 1;
            if ((unsigned)gx >= 128u) continue;
            acc += x[((b * 64 + c) * 128 + gy) * 128 + gx] * w[ky * 3 + kx];
        }
    }
    h1[idx] = (_Float16)geluf(acc);
}

// ------------- stage 2: dense conv3x3 C->C as implicit GEMM (WMMA) ---------
// block = 256 (8 waves) = 4 M-tiles x 2 groups of 16 pixels (32 px per block)
__global__ void __launch_bounds__(256)
k_fgconv(const _Float16* __restrict__ h1, const _Float16* __restrict__ wpC,
         _Float16* __restrict__ h2) {
    __shared__ __align__(16) _Float16 tile[3 * 34 * 64];   // [row][col][cin] f16
    int tid = threadIdx.x;
    int g = blockIdx.x;                                    // 4*128*4 = 2048
    int b = g >> 9, rem = g & 511, y = rem >> 2, x0 = (rem & 3) * 32;
    for (int i = tid; i < 3 * 34 * 64; i += 256) {
        int cin = i & 63, col = (i >> 6) % 34, row = i / (34 * 64);
        int gy = y + row - 1, gx = x0 + col - 1;
        _Float16 v = (_Float16)0.0f;                       // SAME = zero pad
        if ((unsigned)gy < 128u && (unsigned)gx < 128u)
            v = h1[((b * 64 + cin) * 128 + gy) * 128 + gx];
        tile[i] = v;
    }
    __syncthreads();

    int wave = tid >> 5, lane = tid & 31;
    int mt = wave & 3, grp = wave >> 2;
    int n = lane & 15, hi = lane >> 4;
    v8f acc = {};
    const v16h* A = (const v16h*)wpC + (mt * 18 * 32 + lane);
#pragma unroll
    for (int kc = 0; kc < 18; ++kc) {                      // K = 9 taps * 64 cin
        v16h a = A[kc * 32];
        int tap = kc >> 1, cb = (kc & 1) * 32;
        int dy = tap / 3, dx = tap % 3;
        int base = (dy * 34 + (grp * 16 + n + dx)) * 64 + cb + 8 * hi;
        h8 lo = *(const h8*)(tile + base);                 // 16B-aligned ds_load_b128
        h8 hh = *(const h8*)(tile + base + 16);
        v16h bf;
#pragma unroll
        for (int j = 0; j < 8; ++j) { bf[j] = lo[j]; bf[j + 8] = hh[j]; }
        acc = WMMA_F16(a, bf, acc);
    }
    int px = x0 + grp * 16 + n;
#pragma unroll
    for (int r = 0; r < 8; ++r) {
        int c = mt * 16 + r + 8 * hi;
        h2[((b * 64 + c) * 128 + y) * 128 + px] = (_Float16)geluf(acc[r]);
    }
}

// -------- stage 3: 1x1 proj C->400 (WMMA) + fused softmax over 25 taps -----
// one wave = one (16-pixel tile, phase); 2 M-tiles cover the 25 taps
__global__ void __launch_bounds__(256)
k_proj_softmax(const _Float16* __restrict__ h2, const _Float16* __restrict__ wpP,
               _Float16* __restrict__ filt) {
    int tid = threadIdx.x, wave = tid >> 5, lane = tid & 31;
    int wid = blockIdx.x * 8 + wave;                       // 65536 waves
    int T = wid >> 4, s = wid & 15;
    int b = T >> 10, rem = T & 1023, y = rem >> 3, x0 = (rem & 7) * 16;
    int n = lane & 15, hi = lane >> 4, px = x0 + n;

    v8f acc0 = {}, acc1 = {};
    const v16h* W = (const v16h*)wpP;
#pragma unroll
    for (int kc = 0; kc < 2; ++kc) {
        v16h bf;
#pragma unroll
        for (int j = 0; j < 16; ++j) {
            int kl = ((j < 8) ? j : j + 8) + 8 * hi;
            int cin = kc * 32 + kl;
            bf[j] = h2[((b * 64 + cin) * 128 + y) * 128 + px];
        }
        v16h a0 = W[((s * 2 + 0) * 2 + kc) * 32 + lane];
        v16h a1 = W[((s * 2 + 1) * 2 + kc) * 32 + lane];
        acc0 = WMMA_F16(a0, bf, acc0);
        acc1 = WMMA_F16(a1, bf, acc1);
    }
    // softmax over taps 0..24; column N lives in lanes N and N+16
    float pmax = -1e30f;
#pragma unroll
    for (int r = 0; r < 8; ++r) {
        pmax = fmaxf(pmax, acc0[r]);
        if (r + 8 * hi < 9) pmax = fmaxf(pmax, acc1[r]);
    }
    float m = fmaxf(pmax, __shfl_xor(pmax, 16, 32));
    float e0[8], e1[8], ps = 0.0f;
#pragma unroll
    for (int r = 0; r < 8; ++r) {
        e0[r] = __expf(acc0[r] - m);
        ps += e0[r];
        if (r + 8 * hi < 9) { e1[r] = __expf(acc1[r] - m); ps += e1[r]; }
        else e1[r] = 0.0f;
    }
    float inv = 1.0f / (ps + __shfl_xor(ps, 16, 32));
    int pix = y * 128 + px;
    _Float16* fp = filt + (b * 16 + s) * 25 * 16384 + pix;
#pragma unroll
    for (int r = 0; r < 8; ++r) {
        fp[(r + 8 * hi) * 16384] = (_Float16)(e0[r] * inv);
        if (r + 8 * hi < 9) fp[(16 + r + 8 * hi) * 16384] = (_Float16)(e1[r] * inv);
    }
}

// ------------------------- gate branch (tiny) ------------------------------
__global__ void k_gate(const float* __restrict__ x, const float* __restrict__ w1,
                       const float* __restrict__ w2, const float* __restrict__ b2,
                       float* __restrict__ gate) {
    int idx = blockIdx.x * 256 + threadIdx.x;              // 65536 exact
    int xx = idx & 127, yy = (idx >> 7) & 127, b = idx >> 14;
    float xv[64];
#pragma unroll
    for (int i = 0; i < 64; ++i) xv[i] = x[((b * 64 + i) * 128 + yy) * 128 + xx];
    float sacc = 0.0f;
    for (int j = 0; j < 16; ++j) {
        float a = 0.0f;
#pragma unroll
        for (int i = 0; i < 64; ++i) a += w1[j * 64 + i] * xv[i];
        a = (a >= 0.0f) ? a : 0.2f * a;                    // LeakyReLU(0.2)
        sacc += w2[j] * a;
    }
    gate[idx] = 1.0f / (1.0f + __expf(-(sacc + b2[0])));
}

// --------- mlp1 feature part: m1x = W1[:, :64] @ x (once per LR pixel) -----
__global__ void __launch_bounds__(256)
k_mlp1(const _Float16* __restrict__ xh, const _Float16* __restrict__ wpM1,
       float* __restrict__ m1x) {
    int tid = threadIdx.x, wave = tid >> 5, lane = tid & 31;
    int wid = blockIdx.x * 8 + wave;                       // 16384 waves
    int T = wid >> 2, mt = wid & 3;
    int b = T >> 10, rem = T & 1023, y = rem >> 3, x0 = (rem & 7) * 16;
    int n = lane & 15, hi = lane >> 4, px = x0 + n;
    v8f acc = {};
#pragma unroll
    for (int kc = 0; kc < 2; ++kc) {
        v16h bf;
#pragma unroll
        for (int j = 0; j < 16; ++j) {
            int kl = ((j < 8) ? j : j + 8) + 8 * hi;
            int cin = kc * 32 + kl;
            bf[j] = xh[((b * 64 + cin) * 128 + y) * 128 + px];
        }
        v16h a = ((const v16h*)wpM1)[(mt * 2 + kc) * 32 + lane];
        acc = WMMA_F16(a, bf, acc);
    }
#pragma unroll
    for (int r = 0; r < 8; ++r) {
        int c = mt * 16 + r + 8 * hi;
        m1x[((b * 64 + c) * 128 + y) * 128 + px] = acc[r];
    }
}

// ----- final: dyn filtering + mlp2 (WMMA) + gate + pixel-shuffle store -----
// block = 512 (16 waves) = one 16-pixel LR tile; wave w handles phase w
__global__ void __launch_bounds__(512)
k_combine(const float* __restrict__ x, const _Float16* __restrict__ filt,
          const float* __restrict__ m1x, const float* __restrict__ gate,
          const _Float16* __restrict__ wpM2, const float* __restrict__ w1,
          const float* __restrict__ b1, const float* __restrict__ b2,
          float* __restrict__ out) {
    __shared__ float xt[64 * 5 * 20];                      // [c][row][col], reflect halo
    int tid = threadIdx.x;
    int T = blockIdx.x;                                    // 4096 tiles
    int b = T >> 10, rem = T & 1023, y = rem >> 3, x0 = (rem & 7) * 16;
    for (int i = tid; i < 6400; i += 512) {
        int col = i % 20, row = (i / 20) % 5, c = i / 100;
        int gy = refl128(y + row - 2), gx = refl128(x0 + col - 2);
        xt[i] = x[((b * 64 + c) * 128 + gy) * 128 + gx];
    }
    __syncthreads();

    int wave = tid >> 5, lane = tid & 31;
    int s = wave, sy = s >> 2, sx = s & 3;
    float gxc = -0.75f + 0.5f * (float)sx;                 // grid ch0 = xx
    float gyc = -0.75f + 0.5f * (float)sy;                 // grid ch1 = yy
    int n = lane & 15, hi = lane >> 4, px = x0 + n;

    float g = gate[(b << 14) + y * 128 + px];
    float fv[25];
    {
        const _Float16* fp = filt + (b * 16 + s) * 25 * 16384 + y * 128 + px;
#pragma unroll
        for (int k = 0; k < 25; ++k) fv[k] = (float)fp[k * 16384];
    }

    v8f zero = {};
    v8f racc[4] = {zero, zero, zero, zero};
#pragma unroll
    for (int kc = 0; kc < 2; ++kc) {
        v16h bf;
#pragma unroll
        for (int j = 0; j < 16; ++j) {
            int kl = ((j < 8) ? j : j + 8) + 8 * hi;
            int cin = kc * 32 + kl;
            float z = m1x[((b * 64 + cin) * 128 + y) * 128 + px]
                    + w1[cin * 66 + 64] * gxc + w1[cin * 66 + 65] * gyc + b1[cin];
            bf[j] = (_Float16)(z > 0.0f ? z : 0.0f);       // ReLU
        }
#pragma unroll
        for (int t = 0; t < 4; ++t) {
            v16h a = ((const v16h*)wpM2)[(t * 2 + kc) * 32 + lane];
            racc[t] = WMMA_F16(a, bf, racc[t]);
        }
    }

#pragma unroll
    for (int t = 0; t < 4; ++t) {
#pragma unroll
        for (int r = 0; r < 8; ++r) {
            int c = t * 16 + r + 8 * hi;                   // fragment row -> channel
            const float* xc = xt + c * 100;
            float dyn = 0.0f;
#pragma unroll
            for (int ky = 0; ky < 5; ++ky)
#pragma unroll
                for (int kx = 0; kx < 5; ++kx)
                    dyn += xc[ky * 20 + n + kx] * fv[ky * 5 + kx];
            float rv = racc[t][r] + b2[c];
            out[((size_t)(b * 64 + c) * 512 + (y * 4 + sy)) * 512 + (px * 4 + sx)]
                = dyn + rv * g;
        }
    }
}

// ---------------------------------------------------------------------------
extern "C" void kernel_launch(void* const* d_in, const int* in_sizes, int n_in,
                              void* d_out, int out_size, void* d_ws, size_t ws_size,
                              hipStream_t stream) {
    (void)in_sizes; (void)n_in; (void)out_size; (void)ws_size;
    const float* x       = (const float*)d_in[0];
    const float* fg_dw   = (const float*)d_in[1];
    const float* fg_conv = (const float*)d_in[2];
    const float* fg_proj = (const float*)d_in[3];
    const float* gate_w1 = (const float*)d_in[4];
    const float* gate_w2 = (const float*)d_in[5];
    const float* gate_b2 = (const float*)d_in[6];
    const float* mlp_w1  = (const float*)d_in[7];
    const float* mlp_b1  = (const float*)d_in[8];
    const float* mlp_w2  = (const float*)d_in[9];
    const float* mlp_b2  = (const float*)d_in[10];

    char* ws = (char*)d_ws;
    _Float16* h1   = (_Float16*)(ws + OFF_H1);
    _Float16* h2   = (_Float16*)(ws + OFF_H2);
    _Float16* xh   = (_Float16*)(ws + OFF_XH);
    _Float16* filt = (_Float16*)(ws + OFF_FILT);
    float*    m1x  = (float*)(ws + OFF_M1X);
    float*    gbuf = (float*)(ws + OFF_GATE);
    _Float16* wpC  = (_Float16*)(ws + OFF_WPC);
    _Float16* wpP  = (_Float16*)(ws + OFF_WPP);
    _Float16* wpM1 = (_Float16*)(ws + OFF_WPM1);
    _Float16* wpM2 = (_Float16*)(ws + OFF_WPM2);

    k_pack_conv<<<144, 256, 0, stream>>>(fg_conv, wpC);
    k_pack_proj<<<128, 256, 0, stream>>>(fg_proj, wpP);
    k_pack64<<<16, 256, 0, stream>>>(mlp_w1, 66, wpM1);
    k_pack64<<<16, 256, 0, stream>>>(mlp_w2, 64, wpM2);
    k_x2h<<<16384, 256, 0, stream>>>(x, xh);

    k_dw_gelu<<<16384, 256, 0, stream>>>(x, fg_dw, h1);
    k_fgconv<<<2048, 256, 0, stream>>>(h1, wpC, h2);
    k_proj_softmax<<<8192, 256, 0, stream>>>(h2, wpP, filt);
    k_gate<<<256, 256, 0, stream>>>(x, gate_w1, gate_w2, gate_b2, gbuf);
    k_mlp1<<<2048, 256, 0, stream>>>(xh, wpM1, m1x);
    k_combine<<<4096, 512, 0, stream>>>(x, filt, m1x, gbuf, wpM2,
                                        mlp_w1, mlp_b1, mlp_b2, (float*)d_out);
}
